// BiGNN_53772990546511
// MI455X (gfx1250) — matrix operand order repaired and checked by reading, not compiled, not run on hardware
//
#include <hip/hip_runtime.h>
#include <hip/hip_bf16.h>

#define ALPHA 0.2f

constexpr int B_ = 4, D_ = 5, E_ = 4096, DM = 256, HD = 256;
constexpr int NU = 1024, NL = 1024, NN = NU + 2 * NL; // 3072

typedef __attribute__((ext_vector_type(16))) __bf16 v16bf;
typedef __attribute__((ext_vector_type(8)))  float  v8f;

// ---- monotonic float <-> uint encoding for atomicMax on floats ----
__device__ __forceinline__ unsigned fenc(float f) {
    unsigned u = __float_as_uint(f);
    return (u & 0x80000000u) ? ~u : (u | 0x80000000u);
}
__device__ __forceinline__ float fdec(unsigned e) {
    unsigned u = (e & 0x80000000u) ? (e & 0x7fffffffu) : ~e;
    return __uint_as_float(u);
}

__global__ void zero_f32(float* __restrict__ p, size_t n) {
    size_t i = (size_t)blockIdx.x * blockDim.x + threadIdx.x;
    if (i < n) p[i] = 0.0f;
}

// Wt[col][k] = bf16(W[k][col]) so WMMA B-fragments are contiguous loads.
__global__ void convert_W(const float* __restrict__ W, __bf16* __restrict__ Wt) {
    int col = blockIdx.x, k = threadIdx.x;
    Wt[(size_t)col * DM + k] = (__bf16)W[(size_t)k * HD + col];
}

// ---------------- user embedding (day 0, both link types) ----------------
__global__ void scatter_user(const int* __restrict__ mob, const int* __restrict__ text,
                             const float* __restrict__ x_loc,
                             float* __restrict__ sums, float* __restrict__ cnt) {
    int b = blockIdx.x / (2 * E_), i = blockIdx.x % (2 * E_);
    const int* p = (i < E_) ? mob  + ((size_t)(b * D_) * E_ + i) * 2
                            : text + ((size_t)(b * D_) * E_ + (i - E_)) * 2;
    int u = p[0], l = p[1];
    int ch = threadIdx.x;
    atomicAdd(&sums[((size_t)b * NU + u) * DM + ch], x_loc[(size_t)l * DM + ch]);
    if (ch == 0) atomicAdd(&cnt[b * NU + u], 1.0f);
}

__global__ void batch_mean(const float* __restrict__ sums, const float* __restrict__ cnt,
                           float* __restrict__ meanv) {
    int b = blockIdx.x, ch = threadIdx.x;
    float s = 0.0f, nwith = 0.0f;
    for (int u = 0; u < NU; ++u) {
        float c = cnt[b * NU + u];
        if (c > 0.0f) {
            s += sums[((size_t)b * NU + u) * DM + ch] / c;
            nwith += 1.0f;
        }
    }
    meanv[b * DM + ch] = s / fmaxf(nwith, 1.0f);
}

__global__ void fill_h0(const float* __restrict__ sums, const float* __restrict__ cnt,
                        const float* __restrict__ meanv, const float* __restrict__ x_loc,
                        float* __restrict__ outbase) {
    int n = blockIdx.x % NN, b = blockIdx.x / NN, ch = threadIdx.x;
    float val;
    if (n < NU) {
        float c = cnt[b * NU + n];
        val = (c > 0.0f) ? sums[((size_t)b * NU + n) * DM + ch] / c : meanv[b * DM + ch];
    } else {
        int l = (n - NU) % NL;
        val = x_loc[(size_t)l * DM + ch];
    }
    outbase[((size_t)(b * D_) * NN + n) * (size_t)HD + ch] = val;
}

// ---------------- Wh = h @ W via bf16 WMMA, f32 accumulate ----------------
// Workgroup = 8 waves; all share one 64-column B tile, async-staged into LDS.
// Each wave computes a 16(M) x 64(N) tile: 4 accumulators, K loop of 8 WMMAs.
__global__ __launch_bounds__(256) void gemm_wh(const float* __restrict__ hbase, int day,
                                               const __bf16* __restrict__ Wt,
                                               float* __restrict__ Wh) {
    __shared__ __align__(32) __bf16 Bs[64 * DM];  // 32 KB: 64 cols x K=256

    int wave = threadIdx.x >> 5;
    int lane = threadIdx.x & 31;
    int lmod = lane & 15, lhalf = lane >> 4;
    int mt   = blockIdx.x * 8 + wave;     // 0..191 (row tile of 16)
    int col0 = blockIdx.y * 64;
    int b    = blockIdx.z;

    // ---- async stage: 32 KB contiguous B tile (Wt is col-major) -> LDS ----
    {
        const char* gsrc = (const char*)(Wt + (size_t)col0 * DM);
        unsigned ldsbase = (unsigned)(size_t)(&Bs[0]);   // low 32 bits = LDS offset
        unsigned toff = (unsigned)threadIdx.x * 16u;
#pragma unroll
        for (int c = 0; c < 8; ++c) {
            unsigned o = toff + (unsigned)c * 4096u;
            unsigned lp = ldsbase + o;
            unsigned long long gp = (unsigned long long)(gsrc + o);
            asm volatile("global_load_async_to_lds_b128 %0, %1, off"
                         :: "v"(lp), "v"(gp) : "memory");
        }
        asm volatile("s_wait_asynccnt 0" ::: "memory");
        __syncthreads();
    }

    const float* hrow = hbase + ((size_t)(b * D_ + day) * NN + mt * 16 + lmod) * (size_t)DM;

    v8f acc0 = {}, acc1 = {}, acc2 = {}, acc3 = {};
    for (int k0 = 0; k0 < DM; k0 += 32) {
        // A fragment (16x32 bf16): lane half 0 -> K=k0..+15; half 1 -> k0+16..+31
        int ka = k0 + lhalf * 16;
        const float* pa = hrow + ka;
        v16bf af;
#pragma unroll
        for (int j = 0; j < 8; ++j) af[j] = (__bf16)pa[j];
#pragma unroll
        for (int j = 0; j < 8; ++j) af[8 + j] = (__bf16)pa[8 + j];

        // B fragments (32x16 bf16): lane holds one column, 16 contiguous K (from LDS)
        int kb = k0 + lhalf * 16;
        v16bf bf0 = *(const v16bf*)&Bs[( 0 + lmod) * DM + kb];
        v16bf bf1 = *(const v16bf*)&Bs[(16 + lmod) * DM + kb];
        v16bf bf2 = *(const v16bf*)&Bs[(32 + lmod) * DM + kb];
        v16bf bf3 = *(const v16bf*)&Bs[(48 + lmod) * DM + kb];

        acc0 = __builtin_amdgcn_wmma_f32_16x16x32_bf16(false, af, false, bf0, (short)0, acc0, false, false);
        acc1 = __builtin_amdgcn_wmma_f32_16x16x32_bf16(false, af, false, bf1, (short)0, acc1, false, false);
        acc2 = __builtin_amdgcn_wmma_f32_16x16x32_bf16(false, af, false, bf2, (short)0, acc2, false, false);
        acc3 = __builtin_amdgcn_wmma_f32_16x16x32_bf16(false, af, false, bf3, (short)0, acc3, false, false);
    }

    // C/D layout: VGPR r, lanes 0-15 -> M=r, lanes 16-31 -> M=8+r; N = lane%16
    float* o = Wh + ((size_t)b * NN + mt * 16) * (size_t)HD + col0;
#pragma unroll
    for (int r = 0; r < 8; ++r) {
        size_t row = (size_t)(r + 8 * lhalf) * HD;
        o[row +  0 + lmod] = acc0[r];
        o[row + 16 + lmod] = acc1[r];
        o[row + 32 + lmod] = acc2[r];
        o[row + 48 + lmod] = acc3[r];
    }
}

// ---------------- f1 = Wh @ a[:hd], f2 = Wh @ a[hd:] ----------------
__global__ void f1f2_kernel(const float* __restrict__ Wh, const float* __restrict__ a,
                            float* __restrict__ f1, float* __restrict__ f2) {
    int wave = threadIdx.x >> 5, lane = threadIdx.x & 31;
    size_t node = (size_t)blockIdx.x * 8 + wave;  // 0..B*NN-1
    const float* w = Wh + node * HD + lane * 8;
    float s1 = 0.0f, s2 = 0.0f;
#pragma unroll
    for (int j = 0; j < 8; ++j) {
        float v = w[j];
        s1 += v * a[lane * 8 + j];
        s2 += v * a[HD + lane * 8 + j];
    }
    for (int off = 16; off; off >>= 1) {
        s1 += __shfl_xor(s1, off, 32);
        s2 += __shfl_xor(s2, off, 32);
    }
    if (lane == 0) { f1[node] = s1; f2[node] = s2; }
}

// Edge-less rows: uniform softmax -> mean of Wh rows, then relu(elu(x)) == relu(x).
__global__ void colact_kernel(const float* __restrict__ Wh, float* __restrict__ colact) {
    int b = blockIdx.x, ch = threadIdx.x;
    const float* w = Wh + (size_t)b * NN * HD + ch;
    float s = 0.0f;
    for (int n = 0; n < NN; ++n) s += w[(size_t)n * HD];
    colact[b * HD + ch] = fmaxf(s / (float)NN, 0.0f);
}

__device__ __forceinline__ void get_edge(const int* __restrict__ mob,
                                         const int* __restrict__ text,
                                         int b, int day, int i, int& u, int& v) {
    if (i < E_) {
        const int* p = mob + ((size_t)(b * D_ + day) * E_ + i) * 2;
        u = p[0]; v = NU + p[1];
    } else {
        const int* p = text + ((size_t)(b * D_ + day) * E_ + (i - E_)) * 2;
        u = p[0]; v = NU + NL + p[1];
    }
}

__global__ void edge_max(const int* __restrict__ mob, const int* __restrict__ text, int day,
                         const float* __restrict__ f1, const float* __restrict__ f2,
                         unsigned* __restrict__ menc) {
    int gid = blockIdx.x * blockDim.x + threadIdx.x;   // 0..B*2E-1
    int b = gid / (2 * E_), i = gid % (2 * E_);
    int u, v; get_edge(mob, text, b, day, i, u, v);
    float t = f1[(size_t)b * NN + u] + f2[(size_t)b * NN + v];
    float e = (t >= 0.0f) ? t : ALPHA * t;
    atomicMax(&menc[b * NU + u], fenc(e));
}

__global__ void edge_sum(const int* __restrict__ mob, const int* __restrict__ text, int day,
                         const float* __restrict__ f1, const float* __restrict__ f2,
                         const unsigned* __restrict__ menc, const float* __restrict__ Wh,
                         float* __restrict__ numer, float* __restrict__ denom) {
    int b = blockIdx.x / (2 * E_), i = blockIdx.x % (2 * E_);
    int u, v; get_edge(mob, text, b, day, i, u, v);
    float t = f1[(size_t)b * NN + u] + f2[(size_t)b * NN + v];
    float e = (t >= 0.0f) ? t : ALPHA * t;
    float w = __expf(e - fdec(menc[b * NU + u]));
    int ch = threadIdx.x;
    atomicAdd(&numer[((size_t)b * NU + u) * HD + ch],
              w * Wh[((size_t)b * NN + v) * HD + ch]);
    if (ch == 0) atomicAdd(&denom[b * NU + u], w);
}

__global__ void finalize_kernel(const float* __restrict__ numer, const float* __restrict__ denom,
                                const float* __restrict__ colact, float* __restrict__ outbase,
                                int day_out) {
    int n = blockIdx.x % NN, b = blockIdx.x / NN, ch = threadIdx.x;
    float val;
    float d = (n < NU) ? denom[b * NU + n] : 0.0f;
    if (d > 0.0f)
        val = fmaxf(numer[((size_t)b * NU + n) * HD + ch] / d, 0.0f);
    else
        val = colact[b * HD + ch];
    outbase[((size_t)(b * D_ + day_out) * NN + n) * (size_t)HD + ch] = val;
}

extern "C" void kernel_launch(void* const* d_in, const int* in_sizes, int n_in,
                              void* d_out, int out_size, void* d_ws, size_t ws_size,
                              hipStream_t stream) {
    const float* x_loc = (const float*)d_in[0];
    const int*   mob   = (const int*)d_in[1];
    const int*   text  = (const int*)d_in[2];
    const float* W     = (const float*)d_in[3];
    const float* a     = (const float*)d_in[4];
    float* out = (float*)d_out;

    char* ws = (char*)d_ws;
    size_t off = 0;
    auto carve = [&](size_t bytes) -> void* {
        void* p = ws + off;
        off = (off + bytes + 255) & ~(size_t)255;
        return p;
    };
    __bf16*  Wt     = (__bf16*)carve((size_t)DM * HD * 2);
    float*   Wh     = (float*)carve((size_t)B_ * NN * HD * 4);
    float*   f1     = (float*)carve((size_t)B_ * NN * 4);
    float*   f2     = (float*)carve((size_t)B_ * NN * 4);
    float*   colact = (float*)carve((size_t)B_ * HD * 4);
    float*   meanv  = (float*)carve((size_t)B_ * DM * 4);
    float*   cnt    = (float*)carve((size_t)B_ * NU * 4);
    float*   denom  = (float*)carve((size_t)B_ * NU * 4);
    unsigned* menc  = (unsigned*)carve((size_t)B_ * NU * 4);
    float*   accum  = (float*)carve((size_t)B_ * NU * DM * 4); // sums (h0), then numer

    convert_W<<<HD, DM, 0, stream>>>(W, Wt);

    // ---- user embedding & day-0 output ----
    size_t nacc = (size_t)B_ * NU * DM;
    zero_f32<<<(unsigned)((nacc + 255) / 256), 256, 0, stream>>>(accum, nacc);
    zero_f32<<<(B_ * NU + 255) / 256, 256, 0, stream>>>(cnt, (size_t)B_ * NU);
    scatter_user<<<B_ * 2 * E_, DM, 0, stream>>>(mob, text, x_loc, accum, cnt);
    batch_mean<<<B_, DM, 0, stream>>>(accum, cnt, meanv);
    fill_h0<<<B_ * NN, HD, 0, stream>>>(accum, cnt, meanv, x_loc, out);

    // ---- D-1 GAT steps; day t adjacency produces day t+1 ----
    for (int t = 0; t < D_ - 1; ++t) {
        gemm_wh<<<dim3(NN / 16 / 8, HD / 64, B_), 256, 0, stream>>>(out, t, Wt, Wh);
        f1f2_kernel<<<B_ * NN / 8, 256, 0, stream>>>(Wh, a, f1, f2);
        colact_kernel<<<B_, HD, 0, stream>>>(Wh, colact);
        zero_f32<<<(unsigned)((nacc + 255) / 256), 256, 0, stream>>>(accum, nacc);
        zero_f32<<<(B_ * NU + 255) / 256, 256, 0, stream>>>(denom, (size_t)B_ * NU);
        zero_f32<<<(B_ * NU + 255) / 256, 256, 0, stream>>>((float*)menc, (size_t)B_ * NU);
        edge_max<<<B_ * 2 * E_ / 256, 256, 0, stream>>>(mob, text, t, f1, f2, menc);
        edge_sum<<<B_ * 2 * E_, 256, 0, stream>>>(mob, text, t, f1, f2, menc, Wh, accum, denom);
        finalize_kernel<<<B_ * NN, HD, 0, stream>>>(accum, denom, colact, out, t + 1);
    }
}
